// GPT2Block_4423816315327
// MI455X (gfx1250) — compile-verified, matrix operand order checked by reading, not run on hardware
//
#include <hip/hip_runtime.h>
#include <hip/hip_bf16.h>

typedef __attribute__((ext_vector_type(16))) _Float16 v16h;
typedef __attribute__((ext_vector_type(8)))  _Float16 v8h;
typedef __attribute__((ext_vector_type(8)))  float    v8f;

// ---------------------------------------------------------------------------
// WMMA helper (CDNA5 wave32): D = A(16x32 f16) * B(32x16 f16) + C(16x16 f32)
// ---------------------------------------------------------------------------
__device__ __forceinline__ v8f wmma_f16(v16h a, v16h b, v8f c) {
  return __builtin_amdgcn_wmma_f32_16x16x32_f16(
      /*neg_a=*/false, a, /*neg_b=*/false, b,
      /*c_mod=*/(short)0, c, /*reuse_a=*/false, /*reuse_b=*/false);
}

// Load one 16x32 f16 fragment in the CDNA5 A/B register layout.
//   lane L (0..15):  row L,    K = {0..7, 16..23}
//   lane L (16..31): row L-16, K = {8..15, 24..31}
// Works for A tiles (row = M) and B tiles stored as Bt[N][K] (row = N).
__device__ __forceinline__ v16h load_frag16(const _Float16* base, int ld) {
  int lane = threadIdx.x & 31;
  const _Float16* p = base + (size_t)(lane & 15) * ld + ((lane >> 4) << 3);
  v8h lo = *(const v8h*)p;
  v8h hi = *(const v8h*)(p + 16);
  v16h r;
#pragma unroll
  for (int i = 0; i < 8; i++) { r[i] = lo[i]; r[8 + i] = hi[i]; }
  return r;
}

// ---------------------------------------------------------------------------
// Weight convert + transpose: Wt[n][k] = (f16) W[k][n]   (K x N input)
// ---------------------------------------------------------------------------
__global__ void __launch_bounds__(256)
transpose_h(const float* __restrict__ W, _Float16* __restrict__ Wt, int K, int N) {
  __shared__ _Float16 t[32][33];
  int n0 = blockIdx.x * 32, k0 = blockIdx.y * 32;
  int tx = threadIdx.x & 31, ty = threadIdx.x >> 5;  // 32 x 8
#pragma unroll
  for (int r = 0; r < 4; r++)
    t[ty + r * 8][tx] = (_Float16)W[(size_t)(k0 + ty + r * 8) * N + n0 + tx];
  __syncthreads();
#pragma unroll
  for (int r = 0; r < 4; r++)
    Wt[(size_t)(n0 + ty + r * 8) * K + k0 + tx] = t[tx][ty + r * 8];
}

// ---------------------------------------------------------------------------
// LayerNorm over D=1024, output f16 (WMMA operand precision)
// ---------------------------------------------------------------------------
__global__ void __launch_bounds__(256)
ln_rows(const float* __restrict__ x, const float* __restrict__ g,
        const float* __restrict__ bb, _Float16* __restrict__ out) {
  const int D = 1024;
  int row = blockIdx.x;
  int tid = threadIdx.x;
  const float* xr = x + (size_t)row * D;
  float v[4];
  float s = 0.f, s2 = 0.f;
#pragma unroll
  for (int i = 0; i < 4; i++) {
    v[i] = xr[tid + 256 * i];
    s += v[i];
    s2 += v[i] * v[i];
  }
#pragma unroll
  for (int m = 16; m >= 1; m >>= 1) {
    s  += __shfl_xor(s, m, 32);
    s2 += __shfl_xor(s2, m, 32);
  }
  __shared__ float rs[8], rs2[8];
  int wave = tid >> 5, lane = tid & 31;
  if (lane == 0) { rs[wave] = s; rs2[wave] = s2; }
  __syncthreads();
  if (tid == 0) {
    float a = 0.f, b2 = 0.f;
#pragma unroll
    for (int i = 0; i < 8; i++) { a += rs[i]; b2 += rs2[i]; }
    rs[0] = a; rs2[0] = b2;
  }
  __syncthreads();
  s = rs[0]; s2 = rs2[0];
  float mu  = s * (1.f / 1024.f);
  float var = s2 * (1.f / 1024.f) - mu * mu;
  float rsg = rsqrtf(var + 1e-5f);
#pragma unroll
  for (int i = 0; i < 4; i++) {
    int c = tid + 256 * i;
    out[(size_t)row * D + c] = (_Float16)((v[i] - mu) * rsg * g[c] + bb[c]);
  }
}

// ---------------------------------------------------------------------------
// WMMA GEMM with async-to-LDS staged B tile, compile-time epilogues.
//   C[M,N] = A[M,K](f16) * Bt[N,K](f16)^T
// Block = 128 threads (4 waves). Block tile: 256(M) x 64(N), BK = 32.
// Each wave owns 64x64 of C = 4x4 WMMA tiles (16 WMMA / K-step).
// B tile (64x32 f16 = 4KB) is loaded once per block via
// global_load_async_to_lds_b128 (ASYNCcnt), double-buffered.
// ---------------------------------------------------------------------------
enum { EPI_QKV = 0, EPI_RESID_F32 = 1, EPI_BIAS_GELU_H = 2, EPI_BIAS_RESID_F32 = 3 };

__device__ __forceinline__ void issue_b_async(const _Float16* Bt, int K, int n0,
                                              int k0, unsigned lds_base) {
  // 64 rows x 64 bytes = 256 x 16B chunks; 128 threads x 2 chunks.
  unsigned long long sbase = (unsigned long long)(uintptr_t)Bt;
  unsigned c = threadIdx.x * 2u;
#pragma unroll
  for (int i = 0; i < 2; i++, c++) {
    unsigned row = c >> 2, sub = c & 3;
    unsigned voff = ((unsigned)(n0 + row) * (unsigned)K + (unsigned)k0) * 2u + sub * 16u;
    unsigned lds  = lds_base + c * 16u;
    asm volatile("global_load_async_to_lds_b128 %0, %1, %2"
                 :: "v"(lds), "v"(voff), "s"(sbase)
                 : "memory");
  }
}

template <int EPI>
__global__ void __launch_bounds__(128)
gemm_wmma(const _Float16* __restrict__ A, const _Float16* __restrict__ Bt,
          int N, int K,
          const float* __restrict__ bias, const float* __restrict__ resid,
          float* __restrict__ outF, _Float16* __restrict__ outH,
          _Float16* __restrict__ Qh, _Float16* __restrict__ Kh,
          _Float16* __restrict__ Vt) {
  __shared__ __align__(16) _Float16 Bsh[2][64 * 32];
  int wave = threadIdx.x >> 5;
  int lane = threadIdx.x & 31;
  int n0 = blockIdx.x * 64;
  int m0 = blockIdx.y * 256 + wave * 64;

  v8f acc[4][4] = {};

  // prologue: stage first B tile
  issue_b_async(Bt, K, n0, 0, (unsigned)(uintptr_t)&Bsh[0][0]);

  for (int k0 = 0, it = 0; k0 < K; k0 += 32, ++it) {
    int rd = it & 1;
    bool has_next = (k0 + 32 < K);
    if (has_next)
      issue_b_async(Bt, K, n0, k0 + 32, (unsigned)(uintptr_t)&Bsh[1 - rd][0]);

    // A fragments from global (issued before the async wait to overlap)
    v16h af[4];
#pragma unroll
    for (int i = 0; i < 4; i++)
      af[i] = load_frag16(A + (size_t)(m0 + i * 16) * K + k0, K);

    if (has_next) asm volatile("s_wait_asynccnt 0x2" ::: "memory");
    else          asm volatile("s_wait_asynccnt 0x0" ::: "memory");
    __syncthreads();  // B tile `rd` visible to all waves

#pragma unroll
    for (int j = 0; j < 4; j++) {
      v16h bf = load_frag16(&Bsh[rd][j * 16 * 32], 32);
#pragma unroll
      for (int i = 0; i < 4; i++)
        acc[i][j] = wmma_f16(af[i], bf, acc[i][j]);
    }
    __syncthreads();  // everyone done reading `rd` before it is refilled
  }

  int col = lane & 15, rh = lane >> 4;
#pragma unroll
  for (int i = 0; i < 4; i++)
#pragma unroll
    for (int j = 0; j < 4; j++)
#pragma unroll
      for (int v = 0; v < 8; v++) {
        int m = m0 + i * 16 + rh * 8 + v;
        int n = n0 + j * 16 + col;
        float val = acc[i][j][v];
        if constexpr (EPI == EPI_QKV) {
          // N=3072: scatter into per-head layouts; q pre-scaled by 1/sqrt(64)
          int sect = n >> 10;
          int c = n & 1023;
          int h = c >> 6;
          int d = c & 63;
          int b = m >> 11;
          int tt = m & 2047;
          size_t bh = (size_t)(b * 16 + h);
          if (sect == 0)      Qh[(bh * 2048 + tt) * 64 + d] = (_Float16)(val * 0.125f);
          else if (sect == 1) Kh[(bh * 2048 + tt) * 64 + d] = (_Float16)val;
          else                Vt[(bh * 64 + d) * 2048 + tt] = (_Float16)val;
        } else if constexpr (EPI == EPI_RESID_F32) {
          outF[(size_t)m * N + n] = val + resid[(size_t)m * N + n];
        } else if constexpr (EPI == EPI_BIAS_GELU_H) {
          val += bias[n];
          val = 0.5f * val * (1.0f + erff(val * 0.70710678118654752f));
          outH[(size_t)m * N + n] = (_Float16)val;
        } else {  // EPI_BIAS_RESID_F32
          outF[(size_t)m * N + n] = val + bias[n] + resid[(size_t)m * N + n];
        }
      }
}

// ---------------------------------------------------------------------------
// Flash attention (causal). Grid: (T/64, B*H), 128 threads.
// Each wave independently owns a 16-query strip. Key tiles of 128 amortize the
// online-softmax overhead; fragment loads are hoisted ahead of the WMMA chains
// so the scheduler can clause them and stage partial loadcnt waits.
// ---------------------------------------------------------------------------
__global__ void __launch_bounds__(128)
flash_attn(const _Float16* __restrict__ Qh, const _Float16* __restrict__ Kh,
           const _Float16* __restrict__ Vt, _Float16* __restrict__ Oh) {
  const int T = 2048, DH = 64, KT = 128;
  __shared__ __align__(16) _Float16 Pl[4][16 * KT];
  int wave = threadIdx.x >> 5, lane = threadIdx.x & 31;
  int col = lane & 15, rh = lane >> 4;
  int bh = blockIdx.y;               // 0..31
  int b = bh >> 4, h = bh & 15;
  int qb = blockIdx.x * 64;          // query tile base (within batch)
  int qw = qb + wave * 16;           // this wave's query strip

  const _Float16* Qbase = Qh + ((size_t)bh * T + qw) * DH;
  v16h qf0 = load_frag16(Qbase, DH);        // d = 0..31
  v16h qf1 = load_frag16(Qbase + 32, DH);   // d = 32..63

  v8f o[4] = {};
  float mrow[8], lrow[8];
#pragma unroll
  for (int v = 0; v < 8; v++) { mrow[v] = -1e30f; lrow[v] = 0.f; }

  for (int kt = 0; kt <= qw + 15; kt += KT) {
    // S[16 x 128] = Q * K^T  (Q pre-scaled by 1/sqrt(DH)); K-fragments loaded
    // two key-subtiles ahead of their WMMAs.
    v8f s[8];
#pragma unroll
    for (int j = 0; j < 8; j++) s[j] = (v8f){};
#pragma unroll
    for (int jp = 0; jp < 4; jp++) {
      const _Float16* Kb0 = Kh + ((size_t)bh * T + kt + (jp * 2 + 0) * 16) * DH;
      const _Float16* Kb1 = Kh + ((size_t)bh * T + kt + (jp * 2 + 1) * 16) * DH;
      v16h kf[4];
      kf[0] = load_frag16(Kb0, DH);
      kf[1] = load_frag16(Kb0 + 32, DH);
      kf[2] = load_frag16(Kb1, DH);
      kf[3] = load_frag16(Kb1 + 32, DH);
      s[jp * 2 + 0] = wmma_f16(qf0, kf[0], s[jp * 2 + 0]);
      s[jp * 2 + 0] = wmma_f16(qf1, kf[1], s[jp * 2 + 0]);
      s[jp * 2 + 1] = wmma_f16(qf0, kf[2], s[jp * 2 + 1]);
      s[jp * 2 + 1] = wmma_f16(qf1, kf[3], s[jp * 2 + 1]);
    }
    if (kt + KT - 1 > qw) {  // tile reaches past the diagonal: causal mask
#pragma unroll
      for (int j = 0; j < 8; j++)
#pragma unroll
        for (int v = 0; v < 8; v++) {
          int key = kt + j * 16 + col;
          int qr  = qw + rh * 8 + v;
          if (key > qr) s[j][v] = -1e30f;
        }
    }
    // online softmax (rows live across 16 lanes of each half-wave)
    float mt[8], mnew[8], corr[8], ls[8];
#pragma unroll
    for (int v = 0; v < 8; v++) {
      float mm = s[0][v];
#pragma unroll
      for (int j = 1; j < 8; j++) mm = fmaxf(mm, s[j][v]);
#pragma unroll
      for (int d = 1; d <= 8; d <<= 1) mm = fmaxf(mm, __shfl_xor(mm, d, 32));
      mt[v] = mm;
    }
#pragma unroll
    for (int v = 0; v < 8; v++) {
      mnew[v] = fmaxf(mrow[v], mt[v]);
      corr[v] = __expf(mrow[v] - mnew[v]);
      float accp = 0.f;
#pragma unroll
      for (int j = 0; j < 8; j++) {
        float p = __expf(s[j][v] - mnew[v]);
        s[j][v] = p;
        accp += p;
      }
      ls[v] = accp;
    }
#pragma unroll
    for (int v = 0; v < 8; v++) {
#pragma unroll
      for (int d = 1; d <= 8; d <<= 1) ls[v] += __shfl_xor(ls[v], d, 32);
      lrow[v] = lrow[v] * corr[v] + ls[v];
      mrow[v] = mnew[v];
    }
#pragma unroll
    for (int j2 = 0; j2 < 4; j2++)
#pragma unroll
      for (int v = 0; v < 8; v++) o[j2][v] = o[j2][v] * corr[v];

    // P (C-layout) -> LDS -> A-layout fragments (same-wave LDS RAW;
    // compiler inserts s_wait_dscnt, no barrier needed)
    _Float16* pl = &Pl[wave][0];
#pragma unroll
    for (int j = 0; j < 8; j++)
#pragma unroll
      for (int v = 0; v < 8; v++)
        pl[(rh * 8 + v) * KT + j * 16 + col] = (_Float16)s[j][v];

    // O += P * V   (V stored transposed: Vt[bh][d][t]); all fragments for a
    // k-step are loaded before the 4 WMMAs so loads clause together.
#pragma unroll
    for (int ks = 0; ks < 4; ks++) {
      v16h pf = load_frag16(pl + ks * 32, KT);
      v16h vf[4];
#pragma unroll
      for (int j2 = 0; j2 < 4; j2++) {
        const _Float16* Vb = Vt + ((size_t)bh * DH + j2 * 16) * T + kt + ks * 32;
        vf[j2] = load_frag16(Vb, T);
      }
#pragma unroll
      for (int j2 = 0; j2 < 4; j2++)
        o[j2] = wmma_f16(pf, vf[j2], o[j2]);
    }
  }

  // normalize and write concat-head output attn_h[b][t][h*64+d] (f16)
#pragma unroll
  for (int j2 = 0; j2 < 4; j2++)
#pragma unroll
    for (int v = 0; v < 8; v++) {
      int r = rh * 8 + v;
      float val = o[j2][v] / lrow[v];
      Oh[((size_t)b * T + qw + r) * 1024 + h * 64 + j2 * 16 + col] = (_Float16)val;
    }
}

// ---------------------------------------------------------------------------
// Host-side orchestration
// ---------------------------------------------------------------------------
extern "C" void kernel_launch(void* const* d_in, const int* in_sizes, int n_in,
                              void* d_out, int out_size, void* d_ws, size_t ws_size,
                              hipStream_t stream) {
  const float* x    = (const float*)d_in[0];
  const float* ln1g = (const float*)d_in[1];
  const float* ln1b = (const float*)d_in[2];
  const float* Wqkv = (const float*)d_in[3];
  const float* Wout = (const float*)d_in[4];
  const float* ln2g = (const float*)d_in[5];
  const float* ln2b = (const float*)d_in[6];
  const float* W1   = (const float*)d_in[7];
  const float* b1   = (const float*)d_in[8];
  const float* W2   = (const float*)d_in[9];
  const float* b2   = (const float*)d_in[10];
  float* out = (float*)d_out;

  char* ws = (char*)d_ws;
  size_t off = 0;
  auto alloc = [&](size_t bytes) { char* p = ws + off; off += bytes; return p; };

  _Float16* Wqkvt = (_Float16*)alloc(3072ull * 1024 * 2);
  _Float16* Woutt = (_Float16*)alloc(1024ull * 1024 * 2);
  _Float16* W1t   = (_Float16*)alloc(4096ull * 1024 * 2);
  _Float16* W2t   = (_Float16*)alloc(1024ull * 4096 * 2);
  _Float16* xh    = (_Float16*)alloc(4096ull * 1024 * 2);
  _Float16* Qh    = (_Float16*)alloc(4096ull * 1024 * 2);
  _Float16* Kh    = (_Float16*)alloc(4096ull * 1024 * 2);
  _Float16* Vt    = (_Float16*)alloc(4096ull * 1024 * 2);
  _Float16* attnh = (_Float16*)alloc(4096ull * 1024 * 2);
  float*    x1    = (float*)   alloc(4096ull * 1024 * 4);
  _Float16* h2h   = (_Float16*)alloc(4096ull * 1024 * 2);
  _Float16* hh    = (_Float16*)alloc(4096ull * 4096 * 2);
  (void)ws_size; (void)in_sizes; (void)n_in; (void)out_size;

  dim3 b256(256), b128(128);

  // 1) weights -> f16, transposed [N][K]
  transpose_h<<<dim3(3072 / 32, 1024 / 32), b256, 0, stream>>>(Wqkv, Wqkvt, 1024, 3072);
  transpose_h<<<dim3(1024 / 32, 1024 / 32), b256, 0, stream>>>(Wout, Woutt, 1024, 1024);
  transpose_h<<<dim3(4096 / 32, 1024 / 32), b256, 0, stream>>>(W1, W1t, 1024, 4096);
  transpose_h<<<dim3(1024 / 32, 4096 / 32), b256, 0, stream>>>(W2, W2t, 4096, 1024);

  // 2) LN1 -> f16
  ln_rows<<<4096, b256, 0, stream>>>(x, ln1g, ln1b, xh);

  // 3) QKV GEMM (scatter epilogue into per-head layouts, q pre-scaled 1/8)
  gemm_wmma<EPI_QKV><<<dim3(3072 / 64, 4096 / 256), b128, 0, stream>>>(
      xh, Wqkvt, 3072, 1024, nullptr, nullptr, nullptr, nullptr, Qh, Kh, Vt);

  // 4) causal flash attention -> attnh (f16, heads concatenated)
  flash_attn<<<dim3(2048 / 64, 32), b128, 0, stream>>>(Qh, Kh, Vt, attnh);

  // 5) out projection + residual(x) -> x1 (f32)
  gemm_wmma<EPI_RESID_F32><<<dim3(1024 / 64, 4096 / 256), b128, 0, stream>>>(
      attnh, Woutt, 1024, 1024, nullptr, x, x1, nullptr, nullptr, nullptr, nullptr);

  // 6) LN2 -> f16
  ln_rows<<<4096, b256, 0, stream>>>(x1, ln2g, ln2b, h2h);

  // 7) FF1 + bias + exact GELU -> hh (f16)
  gemm_wmma<EPI_BIAS_GELU_H><<<dim3(4096 / 64, 4096 / 256), b128, 0, stream>>>(
      h2h, W1t, 4096, 1024, b1, nullptr, nullptr, hh, nullptr, nullptr, nullptr);

  // 8) FF2 + bias + residual(x1) -> out (f32)
  gemm_wmma<EPI_BIAS_RESID_F32><<<dim3(1024 / 64, 4096 / 256), b128, 0, stream>>>(
      hh, W2t, 1024, 4096, b2, x1, out, nullptr, nullptr, nullptr, nullptr);
}
